// MessageFunction_32504312496663
// MI455X (gfx1250) — matrix operand order repaired and checked by reading, not compiled, not run on hardware
//
#include <hip/hip_runtime.h>

typedef __attribute__((ext_vector_type(16))) __bf16 v16bf;
typedef __attribute__((ext_vector_type(8)))  float  v8f;
typedef __attribute__((ext_vector_type(4)))  int    v4i;

#define E_TOTAL 131072
#define TILE    128
#define STRA    264   // 256 + 8 pad (bf16 elems)
#define STRB    136   // 128 + 8 pad

#if defined(__gfx1250__) && __has_builtin(__builtin_amdgcn_global_load_async_to_lds_b128)
#define HAVE_ASYNC_LDS 1
typedef __attribute__((address_space(1))) v4i gv4i;   // global-space b128 chunk
typedef __attribute__((address_space(3))) v4i lv4i;   // LDS-space b128 chunk
#else
#define HAVE_ASYNC_LDS 0
#endif

// fp32 -> bf16 via native convert (v_cvt_*bf16_f32, RNE); store the raw bits.
__device__ __forceinline__ unsigned short f2bf(float f) {
  return __builtin_bit_cast(unsigned short, (__bf16)f);
}

union Frag16 { uint4 u[2]; v16bf v; };

// ============================================================================
// Prologue: build bf16 fragment-major weights in workspace.
//   B-fragment layout (per 32-K x 16-N tile):
//     lane l holds column n = nt*16 + (l&15), k = kt*32 + (l>>4)*16 + j, j=0..15
//   Segments (bf16 elems): W1f @0 (4096), W2f @4096 (32768), W3f @36864 (32768),
//   W4f(summed over IN) @69632 (4096); bsum4 (32 f32) after 73728 bf16.
// ============================================================================
__global__ void prep_weights(const float* __restrict__ W1, const float* __restrict__ W2,
                             const float* __restrict__ W3, const float* __restrict__ W4,
                             const float* __restrict__ b4,
                             unsigned short* __restrict__ wf, float* __restrict__ bsum) {
  int t = blockIdx.x * blockDim.x + threadIdx.x;
  if (t < 4096) {                       // W1: K=16 (zero-pad to 32), N=128, nKt=1
    int j = t & 15, l = (t >> 4) & 31, nt = t >> 9;
    int k = ((l >> 4) << 4) + j;
    int n = (nt << 4) + (l & 15);
    float v = (k < 16) ? W1[k * 128 + n] : 0.0f;
    wf[t] = f2bf(v);
  } else if (t < 36864) {               // W2: K=128, N=256, nKt=4
    int t2 = t - 4096;
    int j = t2 & 15, l = (t2 >> 4) & 31, q = t2 >> 9, kt = q & 3, nt = q >> 2;
    int k = (kt << 5) + ((l >> 4) << 4) + j;
    int n = (nt << 4) + (l & 15);
    wf[t] = f2bf(W2[k * 256 + n]);
  } else if (t < 69632) {               // W3: K=256, N=128, nKt=8
    int t3 = t - 36864;
    int j = t3 & 15, l = (t3 >> 4) & 31, q = t3 >> 9, kt = q & 7, nt = q >> 3;
    int k = (kt << 5) + ((l >> 4) << 4) + j;
    int n = (nt << 4) + (l & 15);
    wf[t] = f2bf(W3[k * 128 + n]);
  } else if (t < 73728) {               // Wsum4[h,o] = sum_i W4[h, o*32+i]; K=128, N=32, nKt=4
    int t4 = t - 69632;
    int j = t4 & 15, l = (t4 >> 4) & 31, q = t4 >> 9, kt = q & 3, nt = q >> 2;
    int k = (kt << 5) + ((l >> 4) << 4) + j;
    int n = (nt << 4) + (l & 15);
    float s = 0.0f;
    #pragma unroll
    for (int i = 0; i < 32; i++) s += W4[(size_t)k * 1024 + n * 32 + i];
    wf[t] = f2bf(s);
  } else if (t < 73760) {               // bsum4[o] = sum_i b4[o*32+i]
    int o = t - 73728;
    float s = 0.0f;
    #pragma unroll
    for (int i = 0; i < 32; i++) s += b4[o * 32 + i];
    bsum[o] = s;
  }
}

// ============================================================================
// Fused per-tile MLP: 128 edges/block, 8 wave32s, each wave owns a 16-row M-tile.
//   X0(16,pad32) -> X1(128) -> X2(256) -> X3(128) -> Y(32) * s_e -> out
// ============================================================================
__global__ __launch_bounds__(256) void mpnn_fused(
    const float* __restrict__ e_vw, const float* __restrict__ h_w,
    const float* __restrict__ b1, const float* __restrict__ b2,
    const float* __restrict__ b3,
    const unsigned short* __restrict__ wf, const float* __restrict__ bsum4,
    float* __restrict__ out) {
  __shared__ __align__(16) unsigned short bufA[TILE * STRA];  // X0 / X2
  __shared__ __align__(16) unsigned short bufB[TILE * STRB];  // X1 / X3
#if HAVE_ASYNC_LDS
  __shared__ __align__(16) float stage0[TILE * 16];           // raw f32 e_vw tile
#endif

  const int wg   = blockIdx.x;
  const int t    = threadIdx.x;
  const int lane = t & 31;
  const int wave = t >> 5;          // 0..7 == M-tile
  const int mrow = wave << 4;
  const int hi8  = (lane >> 4) << 3;
  const int nlan = lane & 15;

  const unsigned short* w1f = wf;
  const unsigned short* w2f = wf + 4096;
  const unsigned short* w3f = wf + 36864;
  const unsigned short* w4f = wf + 69632;

  // Warm WGP$ with the front of the fragment buffer (global_prefetch_b8).
  __builtin_prefetch((const char*)wf + (t << 6), 0, 0);

  // A-fragment from LDS: lane l -> row mrow + (l&15); K chunks {base, base+8},
  // base = kt*32 + 16*(l>=16). Two 16B loads.
  auto load_a = [&](const unsigned short* buf, int stride, int kt) -> v16bf {
    const unsigned short* p = buf + (mrow + nlan) * stride + (kt << 5) + (hi8 << 1);
    Frag16 f;
    f.u[0] = *(const uint4*)(p);
    f.u[1] = *(const uint4*)(p + 8);
    return f.v;
  };
  // B-fragment from global (L2-resident): one contiguous 32B load per lane.
  auto load_b = [&](const unsigned short* seg, int nt, int kt, int nKt) -> v16bf {
    return *(const v16bf*)(seg + (size_t)(((nt * nKt + kt) << 5) + lane) * 16);
  };

  // ---- Stage 0: e_vw tile (128x16 f32) -> bufA bf16, K padded to 32 with zeros
#if HAVE_ASYNC_LDS
  {
    // ASYNCcnt path: copy raw f32 tile to LDS without touching VGPRs.
    // 8192 B = 512 x b128; 256 threads x 2 chunks each.
    #pragma unroll
    for (int c = 0; c < 2; c++) {
      int idx = t + c * 256;                       // b128 chunk id
      const float* src = e_vw + (size_t)wg * (TILE * 16) + idx * 4;
      float* dst = stage0 + idx * 4;
      __builtin_amdgcn_global_load_async_to_lds_b128((gv4i*)src, (lv4i*)dst, 0, 0);
    }
    asm volatile("s_wait_asynccnt 0x0" ::: "memory");
    __syncthreads();
    int row = t >> 1, half = t & 1;
    const float* src = stage0 + row * 16 + half * 8;
    unsigned short* dst = bufA + row * STRA + half * 8;
    #pragma unroll
    for (int j = 0; j < 8; j++) { dst[j] = f2bf(src[j]); dst[16 + j] = 0; }
  }
#else
  {
    int row = t >> 1, half = t & 1;
    const float* src = e_vw + (size_t)(wg * TILE + row) * 16 + half * 8;
    unsigned short* dst = bufA + row * STRA + half * 8;
    #pragma unroll
    for (int j = 0; j < 8; j++) { dst[j] = f2bf(src[j]); dst[16 + j] = 0; }
  }
#endif
  __syncthreads();

  // ---- Stage 1: X1 = relu(X0 @ W1 + b1)   K=32(pad), N=128 -> bufB
  {
    v16bf a0 = load_a(bufA, STRA, 0);
    #pragma unroll
    for (int nt = 0; nt < 8; nt++) {
      v8f acc = {};
      v16bf b = load_b(w1f, nt, 0, 1);
      acc = __builtin_amdgcn_wmma_f32_16x16x32_bf16(false, a0, false, b, (short)0, acc, false, false);
      int n = (nt << 4) + nlan;
      float bias = b1[n];
      #pragma unroll
      for (int r = 0; r < 8; r++) {
        bufB[(mrow + r + hi8) * STRB + n] = f2bf(fmaxf(acc[r] + bias, 0.0f));
      }
    }
  }
  __syncthreads();

  // ---- Stage 2: X2 = relu(X1 @ W2 + b2)   K=128, N=256 -> bufA
  {
    v16bf a[4];
    #pragma unroll
    for (int kt = 0; kt < 4; kt++) a[kt] = load_a(bufB, STRB, kt);
    #pragma unroll
    for (int nt = 0; nt < 16; nt++) {
      v8f acc = {};
      #pragma unroll
      for (int kt = 0; kt < 4; kt++) {
        v16bf b = load_b(w2f, nt, kt, 4);
        acc = __builtin_amdgcn_wmma_f32_16x16x32_bf16(false, a[kt], false, b, (short)0, acc, false, false);
      }
      int n = (nt << 4) + nlan;
      float bias = b2[n];
      #pragma unroll
      for (int r = 0; r < 8; r++) {
        bufA[(mrow + r + hi8) * STRA + n] = f2bf(fmaxf(acc[r] + bias, 0.0f));
      }
    }
  }
  __syncthreads();

  // ---- Stage 3: X3 = relu(X2 @ W3 + b3)   K=256, N=128 -> bufB
  {
    v16bf a[8];
    #pragma unroll
    for (int kt = 0; kt < 8; kt++) a[kt] = load_a(bufA, STRA, kt);
    #pragma unroll
    for (int nt = 0; nt < 8; nt++) {
      v8f acc = {};
      #pragma unroll
      for (int kt = 0; kt < 8; kt++) {
        v16bf b = load_b(w3f, nt, kt, 8);
        acc = __builtin_amdgcn_wmma_f32_16x16x32_bf16(false, a[kt], false, b, (short)0, acc, false, false);
      }
      int n = (nt << 4) + nlan;
      float bias = b3[n];
      #pragma unroll
      for (int r = 0; r < 8; r++) {
        bufB[(mrow + r + hi8) * STRB + n] = f2bf(fmaxf(acc[r] + bias, 0.0f));
      }
    }
  }
  __syncthreads();

  // ---- Stage 4: m[e,o] = s_e * (X3 @ Wsum4 + bsum4)   K=128, N=32 -> out
  {
    v16bf a[4];
    #pragma unroll
    for (int kt = 0; kt < 4; kt++) a[kt] = load_a(bufB, STRB, kt);
    #pragma unroll
    for (int nt = 0; nt < 2; nt++) {
      v8f acc = {};
      #pragma unroll
      for (int kt = 0; kt < 4; kt++) {
        v16bf b = load_b(w4f, nt, kt, 4);
        acc = __builtin_amdgcn_wmma_f32_16x16x32_bf16(false, a[kt], false, b, (short)0, acc, false, false);
      }
      int n = (nt << 4) + nlan;
      float bias = bsum4[n];
      #pragma unroll
      for (int r = 0; r < 8; r++) {
        int m = mrow + r + hi8;
        float s = h_w[wg * 32 + (m >> 2)];   // h_w_rows rows are constant: h_w[e/128, (e%128)/4]
        out[((size_t)(wg * TILE + m)) * 32 + n] = (acc[r] + bias) * s;
      }
    }
  }
}

extern "C" void kernel_launch(void* const* d_in, const int* in_sizes, int n_in,
                              void* d_out, int out_size, void* d_ws, size_t ws_size,
                              hipStream_t stream) {
  (void)in_sizes; (void)n_in; (void)out_size; (void)ws_size;
  // setup_inputs order: h_v, h_w, e_vw, W1, b1, W2, b2, W3, b3, W4, b4 (h_v unused)
  const float* h_w  = (const float*)d_in[1];
  const float* e_vw = (const float*)d_in[2];
  const float* W1   = (const float*)d_in[3];
  const float* b1   = (const float*)d_in[4];
  const float* W2   = (const float*)d_in[5];
  const float* b2   = (const float*)d_in[6];
  const float* W3   = (const float*)d_in[7];
  const float* b3   = (const float*)d_in[8];
  const float* W4   = (const float*)d_in[9];
  const float* b4   = (const float*)d_in[10];

  unsigned short* wf  = (unsigned short*)d_ws;
  float* bsum         = (float*)((char*)d_ws + (size_t)73728 * 2);

  prep_weights<<<(73760 + 255) / 256, 256, 0, stream>>>(W1, W2, W3, W4, b4, wf, bsum);
  mpnn_fused<<<E_TOTAL / TILE, 256, 0, stream>>>(e_vw, h_w, b1, b2, b3, wf, bsum, (float*)d_out);
}
